// G_27401891349039
// MI455X (gfx1250) — matrix operand order repaired and checked by reading, not compiled, not run on hardware
//
#include <hip/hip_runtime.h>
#include <hip/hip_bf16.h>

typedef __attribute__((ext_vector_type(16))) _Float16 v16h;
typedef __attribute__((ext_vector_type(8)))  float    v8f;

#define K_BETA  0.1f
#define K_ALPHA 0.01f

// weight-fragment region bases (fragment = 512 halfs = one 32x16 f16 B tile)
#define F_PW1   0
#define F_PW2   2
#define F_PW3   4
#define F_GW    6
#define F_PW1T  8
#define F_PW2T 10
#define F_PW3T 12
#define F_GWT  14
#define F_MW1  16
#define F_MW2  20
#define F_MW3  28
#define N_FRAGS 156

// Branch-free tanh using the hardware transcendental units:
// tanh(x) = 1 - 2/(exp2(2*log2e*x) + 1);  exp2->inf => 1, exp2->0 => -1.
__device__ __forceinline__ float tanh_fast(float x)
{
    float e = __builtin_amdgcn_exp2f(x * 2.8853900817779268f);  // exp(2x)
    return 1.0f - 2.0f * __builtin_amdgcn_rcpf(e + 1.0f);
}

// ---------------------------------------------------------------------------
// Kernel 1: pack all weights (and required transposes) into f16 WMMA B-frags.
// B-frag layout (contiguous k order used here, not the ISA table order): VGPR i holds k = 2i+h (+16 for lanes>=16),
// n = lane%16.  Stored so lane l reads 16 contiguous halfs (32B) per frag.
// ---------------------------------------------------------------------------
__global__ void pack_weights(const float* __restrict__ pW1, const float* __restrict__ pW2,
                             const float* __restrict__ pW3, const float* __restrict__ gW,
                             const float* __restrict__ mW1, const float* __restrict__ mW2,
                             const float* __restrict__ mW3, _Float16* __restrict__ ws)
{
    int frag = blockIdx.x;          // 0..155
    int t    = threadIdx.x;         // 0..511
    int l    = t >> 4;              // lane 0..31
    int slot = t & 15;              // half index within lane
    int i    = slot >> 1, h = slot & 1;
    int p    = l >> 4;
    int kl   = 2 * i + h + 16 * p;  // k within 32-chunk
    int nl   = l & 15;

    const float* src; int K, N, nt, kc; bool tr = false;
    if (frag < 16) {                 // eight 32x32 matrices, 2 n-tiles each
        int r = frag >> 1; nt = frag & 1; kc = 0; K = 32; N = 32;
        int rs = (r >= 4) ? (r - 4) : r;
        tr = (r >= 4);
        src = (rs == 0) ? pW1 : (rs == 1) ? pW2 : (rs == 2) ? pW3 : gW;
    } else if (frag < 20) {          // mW1: 32x64
        src = mW1; K = 32; N = 64; nt = frag - 16; kc = 0;
    } else if (frag < 28) {          // mW2: 64x64
        src = mW2; K = 64; N = 64; int idx = frag - 20; nt = idx >> 1; kc = idx & 1;
    } else {                         // mW3: 64x1024
        src = mW3; K = 64; N = 1024; int idx = frag - 28; nt = idx >> 1; kc = idx & 1;
    }
    int k = kc * 32 + kl;
    int n = nt * 16 + nl;
    float v = tr ? src[n * K + k]    // transposed 32x32: Bt[k][n] = W[n][k]
                 : src[k * N + n];   // normal: B[k][n] = W[k][n]   (W is [in][out])
    ws[(size_t)frag * 512 + l * 16 + slot] = (_Float16)v;
}

// ---------------------------------------------------------------------------
// Helpers for the fused kernel
// ---------------------------------------------------------------------------
__device__ __forceinline__ v16h ldB(const _Float16* __restrict__ ws, int frag, int lane)
{
    return *reinterpret_cast<const v16h*>(ws + (size_t)frag * 512 + lane * 16);
}

// Build an f16 A-fragment from an f32 LDS activation buffer [16][stride].
// A layout (matches packed B): lane<16 -> K = 0..15 contiguous,
// lane>=16 -> +16.  kofs selects the 32-wide K chunk for K=64 inputs.
__device__ __forceinline__ v16h mkA(const float* __restrict__ buf, int stride,
                                    int m, int p, int kofs)
{
    v16h a;
#pragma unroll
    for (int i = 0; i < 8; ++i) {
        int kb = 2 * i + 16 * p + kofs;
        a[2 * i]     = (_Float16)buf[m * stride + kb];
        a[2 * i + 1] = (_Float16)buf[m * stride + kb + 1];
    }
    return a;
}

// per-wave LDS floats: 6*560 + 2*1072 + 2*512 + (16*1024 halfs = 8192 floats)
#define WAVE_FLOATS 14720

// ---------------------------------------------------------------------------
// Kernel 2: fully fused per-sample pipeline. 1 wave = 16 samples.
// ---------------------------------------------------------------------------
__global__ void __launch_bounds__(128)
fused_kernel(const float* __restrict__ x,
             const float* __restrict__ pb1, const float* __restrict__ pb2,
             const float* __restrict__ pb3,
             const float* __restrict__ mb1, const float* __restrict__ mb2,
             const float* __restrict__ mb3,
             const _Float16* __restrict__ wf,
             float* __restrict__ out)
{
    extern __shared__ float smem[];
    const int lane = threadIdx.x & 31;
    const int wave = threadIdx.x >> 5;
    const int p    = lane >> 4;
    const int ml   = lane & 15;          // sample row (A) / n-local (B,D)

    float* base = smem + wave * WAVE_FLOATS;
    float* XB  = base;                   // [16][35] raw x
    float* H1B = base + 560;             // [16][35] tanh layer 1
    float* H2B = base + 1120;            // [16][35] tanh layer 2
    float* PEB = base + 1680;            // [16][35] pe_vec
    float* TMB = base + 2240;            // [16][35] d2 then d1
    float* GB  = base + 2800;            // [16][35] grad_E
    float* M1B = base + 3360;            // [16][67] MW hidden 1
    float* M2B = base + 4432;            // [16][67] MW hidden 2
    float* TB  = base + 5504;            // [16][32] t = L^T g
    float* WB  = base + 6016;            // [16][32] u - v
    _Float16* MW = (_Float16*)(base + 6528);   // [16][1024] mw (f16)

    const int tile0 = (blockIdx.x * 4 + wave) * 16;

    // stage x into LDS (coalesced)
#pragma unroll
    for (int r = 0; r < 16; ++r)
        XB[r * 35 + lane] = x[(size_t)(tile0 + r) * 32 + lane];
    __syncthreads();

    v16h aX = mkA(XB, 35, ml, p, 0);

    // ---- grad_E forward: h1 = tanh(x pW1 + b1) ----
#pragma unroll
    for (int nt = 0; nt < 2; ++nt) {
        v8f acc = {};
        acc = __builtin_amdgcn_wmma_f32_16x16x32_f16(false, aX, false,
                 ldB(wf, F_PW1 + nt, lane), (short)0, acc, false, false);
        int n = nt * 16 + ml; float b = pb1[n];
#pragma unroll
        for (int i = 0; i < 8; ++i) H1B[(i + 8 * p) * 35 + n] = tanh_fast(acc[i] + b);
    }
    __syncthreads();
    v16h aH1 = mkA(H1B, 35, ml, p, 0);

    // ---- h2 = tanh(h1 pW2 + b2) ----
#pragma unroll
    for (int nt = 0; nt < 2; ++nt) {
        v8f acc = {};
        acc = __builtin_amdgcn_wmma_f32_16x16x32_f16(false, aH1, false,
                 ldB(wf, F_PW2 + nt, lane), (short)0, acc, false, false);
        int n = nt * 16 + ml; float b = pb2[n];
#pragma unroll
        for (int i = 0; i < 8; ++i) H2B[(i + 8 * p) * 35 + n] = tanh_fast(acc[i] + b);
    }
    __syncthreads();
    v16h aH2 = mkA(H2B, 35, ml, p, 0);

    // ---- pe = h2 pW3 + b3 + x gW ----
#pragma unroll
    for (int nt = 0; nt < 2; ++nt) {
        v8f acc = {};
        acc = __builtin_amdgcn_wmma_f32_16x16x32_f16(false, aH2, false,
                 ldB(wf, F_PW3 + nt, lane), (short)0, acc, false, false);
        acc = __builtin_amdgcn_wmma_f32_16x16x32_f16(false, aX, false,
                 ldB(wf, F_GW + nt, lane), (short)0, acc, false, false);
        int n = nt * 16 + ml; float b = pb3[n];
#pragma unroll
        for (int i = 0; i < 8; ++i) PEB[(i + 8 * p) * 35 + n] = acc[i] + b;
    }
    __syncthreads();
    v16h aPE = mkA(PEB, 35, ml, p, 0);

    // ---- d2 = (pe pW3^T) * (1 - h2^2) ----
#pragma unroll
    for (int nt = 0; nt < 2; ++nt) {
        v8f acc = {};
        acc = __builtin_amdgcn_wmma_f32_16x16x32_f16(false, aPE, false,
                 ldB(wf, F_PW3T + nt, lane), (short)0, acc, false, false);
        int n = nt * 16 + ml;
#pragma unroll
        for (int i = 0; i < 8; ++i) {
            float hv = H2B[(i + 8 * p) * 35 + n];
            TMB[(i + 8 * p) * 35 + n] = acc[i] * (1.0f - hv * hv);
        }
    }
    __syncthreads();
    v16h aD2 = mkA(TMB, 35, ml, p, 0);
    __syncthreads();   // WAR: TMB re-used for d1

    // ---- d1 = (d2 pW2^T) * (1 - h1^2) ----
#pragma unroll
    for (int nt = 0; nt < 2; ++nt) {
        v8f acc = {};
        acc = __builtin_amdgcn_wmma_f32_16x16x32_f16(false, aD2, false,
                 ldB(wf, F_PW2T + nt, lane), (short)0, acc, false, false);
        int n = nt * 16 + ml;
#pragma unroll
        for (int i = 0; i < 8; ++i) {
            float hv = H1B[(i + 8 * p) * 35 + n];
            TMB[(i + 8 * p) * 35 + n] = acc[i] * (1.0f - hv * hv);
        }
    }
    __syncthreads();
    v16h aD1 = mkA(TMB, 35, ml, p, 0);

    // ---- g = 2*beta*x + pe gW^T + d1 pW1^T ----
#pragma unroll
    for (int nt = 0; nt < 2; ++nt) {
        v8f acc = {};
        acc = __builtin_amdgcn_wmma_f32_16x16x32_f16(false, aPE, false,
                 ldB(wf, F_GWT + nt, lane), (short)0, acc, false, false);
        acc = __builtin_amdgcn_wmma_f32_16x16x32_f16(false, aD1, false,
                 ldB(wf, F_PW1T + nt, lane), (short)0, acc, false, false);
        int n = nt * 16 + ml;
#pragma unroll
        for (int i = 0; i < 8; ++i) {
            float xv = XB[(i + 8 * p) * 35 + n];
            GB[(i + 8 * p) * 35 + n] = acc[i] + 2.0f * K_BETA * xv;
        }
    }
    __syncthreads();

    // ---- MW_net: m1 = tanh(x mW1 + mb1)  (32 -> 64) ----
#pragma unroll
    for (int nt = 0; nt < 4; ++nt) {
        v8f acc = {};
        acc = __builtin_amdgcn_wmma_f32_16x16x32_f16(false, aX, false,
                 ldB(wf, F_MW1 + nt, lane), (short)0, acc, false, false);
        int n = nt * 16 + ml; float b = mb1[n];
#pragma unroll
        for (int i = 0; i < 8; ++i) M1B[(i + 8 * p) * 67 + n] = tanh_fast(acc[i] + b);
    }
    __syncthreads();
    v16h aM1lo = mkA(M1B, 67, ml, p, 0);
    v16h aM1hi = mkA(M1B, 67, ml, p, 32);

    // ---- m2 = tanh(m1 mW2 + mb2)  (64 -> 64) ----
#pragma unroll
    for (int nt = 0; nt < 4; ++nt) {
        v8f acc = {};
        acc = __builtin_amdgcn_wmma_f32_16x16x32_f16(false, aM1lo, false,
                 ldB(wf, F_MW2 + nt * 2 + 0, lane), (short)0, acc, false, false);
        acc = __builtin_amdgcn_wmma_f32_16x16x32_f16(false, aM1hi, false,
                 ldB(wf, F_MW2 + nt * 2 + 1, lane), (short)0, acc, false, false);
        int n = nt * 16 + ml; float b = mb2[n];
#pragma unroll
        for (int i = 0; i < 8; ++i) M2B[(i + 8 * p) * 67 + n] = tanh_fast(acc[i] + b);
    }
    __syncthreads();
    v16h aM2lo = mkA(M2B, 67, ml, p, 0);
    v16h aM2hi = mkA(M2B, 67, ml, p, 32);

    // ---- mw = m2 mW3 + mb3  (64 -> 1024): the dominant GEMM, 128 WMMAs ----
    for (int nt = 0; nt < 64; ++nt) {
        v8f acc = {};
        acc = __builtin_amdgcn_wmma_f32_16x16x32_f16(false, aM2lo, false,
                 ldB(wf, F_MW3 + nt * 2 + 0, lane), (short)0, acc, false, false);
        acc = __builtin_amdgcn_wmma_f32_16x16x32_f16(false, aM2hi, false,
                 ldB(wf, F_MW3 + nt * 2 + 1, lane), (short)0, acc, false, false);
        int n = nt * 16 + ml; float b = mb3[n];
#pragma unroll
        for (int i = 0; i < 8; ++i)
            MW[(size_t)(i + 8 * p) * 1024 + n] = (_Float16)(acc[i] + b);
    }
    __syncthreads();

    // ---- structured products: t = L^T g, v = U^T g, u = U g ----
#pragma unroll 1
    for (int s = 0; s < 16; ++s) {
        float t = 0.0f, u = 0.0f, v = 0.0f;
        // column access: conflict-free (consecutive lanes -> consecutive banks)
#pragma unroll 1
        for (int row = 0; row < 32; ++row) {
            float m  = (float)MW[(size_t)s * 1024 + row * 32 + lane];
            float gr = GB[s * 35 + row];                  // broadcast
            if (row >= lane) t += m * gr;                 // t_k, k = lane
            if (row <= lane) v += m * gr;                 // (U^T g)_j, j = lane
        }
        // row access with rotation to avoid bank conflicts
#pragma unroll 1
        for (int c = 0; c < 32; ++c) {
            int k = (lane + c) & 31;
            float m  = (float)MW[(size_t)s * 1024 + lane * 32 + k];
            float gk = GB[s * 35 + k];
            if (k >= lane) u += m * gk;                   // (U g)_j, j = lane
        }
        TB[s * 32 + lane] = t;
        WB[s * 32 + lane] = u - v;
    }
    __syncthreads();

    // ---- r = L t ;  out = -(r + u - v) - alpha*g ----
#pragma unroll 1
    for (int s = 0; s < 16; ++s) {
        float r = 0.0f;
#pragma unroll 1
        for (int c = 0; c < 32; ++c) {
            int k = (lane + c) & 31;
            if (k <= lane)
                r += (float)MW[(size_t)s * 1024 + lane * 32 + k] * TB[s * 32 + k];
        }
        float gl = GB[s * 35 + lane];
        out[(size_t)(tile0 + s) * 32 + lane] = -(r + WB[s * 32 + lane]) - K_ALPHA * gl;
    }
}

// ---------------------------------------------------------------------------
extern "C" void kernel_launch(void* const* d_in, const int* in_sizes, int n_in,
                              void* d_out, int out_size, void* d_ws, size_t ws_size,
                              hipStream_t stream)
{
    const float* x   = (const float*)d_in[0];
    const float* pW1 = (const float*)d_in[1];
    const float* pb1 = (const float*)d_in[2];
    const float* pW2 = (const float*)d_in[3];
    const float* pb2 = (const float*)d_in[4];
    const float* pW3 = (const float*)d_in[5];
    const float* pb3 = (const float*)d_in[6];
    const float* gW  = (const float*)d_in[7];
    const float* mW1 = (const float*)d_in[8];
    const float* mb1 = (const float*)d_in[9];
    const float* mW2 = (const float*)d_in[10];
    const float* mb2 = (const float*)d_in[11];
    const float* mW3 = (const float*)d_in[12];
    const float* mb3 = (const float*)d_in[13];
    _Float16* wf = (_Float16*)d_ws;      // 156 KB of packed f16 fragments

    pack_weights<<<N_FRAGS, 512, 0, stream>>>(pW1, pW2, pW3, gW, mW1, mW2, mW3, wf);

    // 65536 samples / (4 waves * 16 samples) = 1024 blocks of 128 threads.
    size_t smem_bytes = (size_t)4 * WAVE_FLOATS * sizeof(float);   // ~230 KB/WG
    fused_kernel<<<1024, 128, smem_bytes, stream>>>(
        x, pb1, pb2, pb3, mb1, mb2, mb3, wf, (float*)d_out);
}
